// MeshGraphNet_3161095929873
// MI455X (gfx1250) — matrix operand order repaired and checked by reading, not compiled, not run on hardware
//
#include <hip/hip_runtime.h>
#include <hip/hip_bf16.h>
#include <stdint.h>

// ---------------------------------------------------------------------------
// CDNA5 (gfx1250) MeshGraphNet: bf16 WMMA 16x16x32 + async/TDM LDS staging.
// Block = 256 threads = 8 wave32; each wave owns a 16x128 output strip.
// ---------------------------------------------------------------------------

typedef __attribute__((ext_vector_type(16))) __bf16 bf16x16;
typedef __attribute__((ext_vector_type(8)))  float  floatx8;
typedef __attribute__((ext_vector_type(4)))  unsigned int uintx4;
typedef __attribute__((ext_vector_type(4)))  int    intx4;
typedef __attribute__((ext_vector_type(8)))  int    intx8;

#define LDIM    128
#define THREADS 256

#if defined(__has_builtin)
#  if __has_builtin(__builtin_amdgcn_tensor_load_to_lds)
#    define HAVE_TDM 1
#  endif
#endif
#ifndef HAVE_TDM
#  define HAVE_TDM 0
#endif

__device__ __forceinline__ unsigned short f2bf(float f) {
  uint32_t u = __builtin_bit_cast(uint32_t, f);
  u += 0x7fffu + ((u >> 16) & 1u);          // round-to-nearest-even
  return (unsigned short)(u >> 16);
}

__device__ __forceinline__ floatx8 fzero8() {
  floatx8 z = {0.f, 0.f, 0.f, 0.f, 0.f, 0.f, 0.f, 0.f};
  return z;
}

// Flat LDS pointer -> 32-bit LDS byte offset (ISA: LDS aperture uses addr[31:0]).
__device__ __forceinline__ unsigned lds_off32(const void* p) {
  return (unsigned)(unsigned long long)p;
}

// Async global -> LDS 16B copy (per-lane addresses, tracked by ASYNCcnt).
__device__ __forceinline__ void async_g2l_b128(unsigned lds_byte, const void* gptr) {
  asm volatile("global_load_async_to_lds_b128 %0, %1, off"
               :: "v"(lds_byte), "v"(gptr)
               : "memory");
}

__device__ __forceinline__ void async_copy_16B(const unsigned short* g,
                                               unsigned short* l, int nchunks) {
  const unsigned base = lds_off32(l);
  for (int i = threadIdx.x; i < nchunks; i += THREADS)
    async_g2l_b128(base + (unsigned)i * 16u, g + (size_t)i * 8);
}

// TDM DMA: contiguous nelem bf16 elements global -> LDS, one descriptor,
// issued by wave 0 only (EXEC-independent, tracked by TENSORcnt).
__device__ __forceinline__ void tdm_stage(const unsigned short* g,
                                          unsigned short* l, unsigned nelem) {
#if HAVE_TDM
  if ((threadIdx.x >> 5) == 0) {
    const unsigned long long ga = (unsigned long long)g;
    uintx4 g0;
    g0[0] = 1u;                                             // count=1, user D#
    g0[1] = lds_off32(l);                                   // lds_addr
    g0[2] = (unsigned)ga;                                   // global_addr lo
    g0[3] = (unsigned)((ga >> 32) & 0x01ffffffu) | (2u << 30); // addr hi | type=2
    intx8 g1;
    g1[0] = 0x10000;                                        // data_size=2B
    g1[1] = (int)((nelem & 0xffffu) << 16);                 // tensor_dim0 lo
    g1[2] = (int)((nelem >> 16) & 0xffffu) | (1 << 16);     // dim0 hi | tensor_dim1=1
    g1[3] = (int)((nelem & 0xffffu) << 16);                 // tile_dim0 = nelem
    g1[4] = 0;                                              // tile_dim1/2 unused
    g1[5] = (int)nelem;                                     // tensor_dim0_stride
    g1[6] = 0;
    g1[7] = 0;
    const intx4 gz4 = {0, 0, 0, 0};
#if defined(__clang_major__) && (__clang_major__ >= 23)
    const intx8 gz8 = {0, 0, 0, 0, 0, 0, 0, 0};
    __builtin_amdgcn_tensor_load_to_lds(g0, g1, gz4, gz4, gz8, 0);
#else
    __builtin_amdgcn_tensor_load_to_lds(g0, g1, gz4, gz4, 0);
#endif
  }
#else
  async_copy_16B(g, l, (int)(nelem >> 3));
#endif
}

// Wait for this wave's async/TDM traffic, then block barrier.
__device__ __forceinline__ void stage_wait() {
  asm volatile("s_wait_asynccnt 0x0" ::: "memory");
#if HAVE_TDM
  __builtin_amdgcn_s_wait_tensorcnt(0);
#endif
  __syncthreads();
}

// A fragment: 16 rows x 32 K slice, CDNA5 16-bit A layout.
__device__ __forceinline__ bf16x16 load_a_frag(const unsigned short* As, int lda, int kbase) {
  const int lane = threadIdx.x & 31;
  const int wave = threadIdx.x >> 5;
  const unsigned short* p = As + (wave * 16 + (lane & 15)) * lda + kbase + ((lane >> 4) << 3);
  bf16x16 a;
#pragma unroll
  for (int e = 0; e < 16; ++e)
    a[e] = __builtin_bit_cast(__bf16, p[(e & 7) + ((e >> 3) << 4)]);
  return a;
}

// B fragment: 32 K x 16 N; lane holds column n = lane%16, K in table order (matches A).
__device__ __forceinline__ bf16x16 load_b_frag(const unsigned short* Ws, int ldn, int kbase, int n0) {
  const int lane = threadIdx.x & 31;
  const unsigned short* p = Ws + (kbase + ((lane >> 4) << 3)) * ldn + n0 + (lane & 15);
  bf16x16 b;
#pragma unroll
  for (int e = 0; e < 16; ++e)
    b[e] = __builtin_bit_cast(__bf16, p[((e & 7) + ((e >> 3) << 4)) * ldn]);
  return b;
}

// 16(rows/wave) x 128(cols) GEMM strip, K multiple of 32.
__device__ __forceinline__ void gemm_tile(const unsigned short* As, int lda,
                                          const unsigned short* Ws, int K,
                                          floatx8* acc) {
  for (int k = 0; k < K; k += 32) {
    bf16x16 a = load_a_frag(As, lda, k);
#pragma unroll
    for (int t = 0; t < 8; ++t) {
      bf16x16 b = load_b_frag(Ws, LDIM, k, t * 16);
      acc[t] = __builtin_amdgcn_wmma_f32_16x16x32_bf16(
          false, a, false, b, (short)0, acc[t], false, false);
    }
  }
}

// bias (+relu) epilogue, result to LDS as bf16 (ld = 128)
__device__ __forceinline__ void epilogue_lds(const floatx8* acc, const float* bias,
                                             bool relu, unsigned short* dst) {
  const int lane = threadIdx.x & 31;
  const int wave = threadIdx.x >> 5;
  const int col0 = lane & 15;
  const int mb = wave * 16 + ((lane >> 4) << 3);
#pragma unroll
  for (int t = 0; t < 8; ++t) {
    const int c = t * 16 + col0;
    const float bv = bias[c];
#pragma unroll
    for (int v = 0; v < 8; ++v) {
      float r = acc[t][v] + bv;
      if (relu) r = fmaxf(r, 0.0f);
      dst[(mb + v) * LDIM + c] = f2bf(r);
    }
  }
}

// ---------------------------------------------------------------------------
// Encoder: [rows, IN_DIM] -> 3-layer MLP -> [rows, 128]
// Weights pre-converted bf16 (W1 pre-padded to 32 x 128).
// LDS: A0(128x128) | A1(128x128) | Wb(128x128)  = 96 KB
// ---------------------------------------------------------------------------
template <int IN_DIM>
__global__ void encoder_kernel(const float* __restrict__ in,
                               const unsigned short* W1p,  // 32 x 128, padded
                               const float* b1,
                               const unsigned short* W2, const float* b2,
                               const unsigned short* W3, const float* b3,
                               float* out_f32, unsigned short* out_bf, int nrows) {
  extern __shared__ unsigned short smem[];
  unsigned short* A0 = smem;
  unsigned short* A1 = smem + LDIM * LDIM;
  unsigned short* Wb = smem + 2 * LDIM * LDIM;
  const int base = blockIdx.x * LDIM;

  // input tile, K padded to 32 (small: stays on the VALU convert path)
  for (int i = threadIdx.x; i < LDIM * 32; i += THREADS) {
    const int r = i >> 5, c = i & 31;
    const int row = base + r;
    float v = 0.0f;
    if (row < nrows && c < IN_DIM) v = in[row * IN_DIM + c];
    A0[r * 32 + c] = f2bf(v);
  }
  tdm_stage(W1p, Wb, 32 * LDIM);
  stage_wait();

  floatx8 acc[8];
#pragma unroll
  for (int t = 0; t < 8; ++t) acc[t] = fzero8();
  gemm_tile(A0, 32, Wb, 32, acc);
  __syncthreads();
  epilogue_lds(acc, b1, true, A1);
  tdm_stage(W2, Wb, LDIM * LDIM);
  stage_wait();

#pragma unroll
  for (int t = 0; t < 8; ++t) acc[t] = fzero8();
  gemm_tile(A1, LDIM, Wb, LDIM, acc);
  __syncthreads();
  epilogue_lds(acc, b2, true, A0);
  tdm_stage(W3, Wb, LDIM * LDIM);
  stage_wait();

#pragma unroll
  for (int t = 0; t < 8; ++t) acc[t] = fzero8();
  gemm_tile(A0, LDIM, Wb, LDIM, acc);

  const int lane = threadIdx.x & 31;
  const int wave = threadIdx.x >> 5;
  const int col0 = lane & 15;
  const int mb = wave * 16 + ((lane >> 4) << 3);
#pragma unroll
  for (int t = 0; t < 8; ++t) {
    const int c = t * 16 + col0;
    const float bv = b3[c];
#pragma unroll
    for (int v = 0; v < 8; ++v) {
      const int row = base + mb + v;
      if (row < nrows) {
        const float r = acc[t][v] + bv;
        if (out_f32) out_f32[row * LDIM + c] = r;
        out_bf[row * LDIM + c] = f2bf(r);
      }
    }
  }
}

// ---------------------------------------------------------------------------
// Edge step: async-gather [h(dst)|h(src)|e] (128x384 bf16) -> 3-layer MLP ->
//            scatter-add f32 messages into agg[dst].
// LDS: A0(128x384) | Wb(384x128) | A1(128x128) | dsts(128) = ~224.5 KB
// ---------------------------------------------------------------------------
__global__ void edge_mlp_kernel(const int* __restrict__ edge_index,
                                const unsigned short* __restrict__ h_bf,
                                const unsigned short* __restrict__ e_bf,
                                const unsigned short* W1, const float* b1,
                                const unsigned short* W2, const float* b2,
                                const unsigned short* W3, const float* b3,
                                float* __restrict__ agg, int nE) {
  extern __shared__ unsigned short smem[];
  unsigned short* A0 = smem;                          // 128*384
  unsigned short* Wb = smem + 128 * 384;              // 384*128
  unsigned short* A1 = Wb + 384 * LDIM;               // 128*128
  int* dsts = (int*)(A1 + LDIM * LDIM);               // 128

  const int ebase = blockIdx.x * LDIM;

  for (int i = threadIdx.x; i < LDIM; i += THREADS) {
    const int e = ebase + i;
    dsts[i] = (e < nE) ? edge_index[nE + e] : -1;
  }

  // async gather: 48 x 16B chunks per edge row, straight into LDS
  const unsigned a0base = lds_off32(A0);
  const uint4 z4 = {0u, 0u, 0u, 0u};
  for (int i = threadIdx.x; i < LDIM * 48; i += THREADS) {
    const int r = i / 48, s = i % 48;
    const int e = ebase + r;
    if (e < nE) {
      const unsigned short* src;
      if (s < 16)      src = h_bf + (size_t)edge_index[nE + e] * LDIM + s * 8;        // x_i
      else if (s < 32) src = h_bf + (size_t)edge_index[e] * LDIM + (s - 16) * 8;      // x_j
      else             src = e_bf + (size_t)e * LDIM + (s - 32) * 8;                  // e
      async_g2l_b128(a0base + (unsigned)i * 16u, src);
    } else {
      ((uint4*)A0)[i] = z4;
    }
  }
  tdm_stage(W1, Wb, 3 * LDIM * LDIM);
  stage_wait();

  floatx8 acc[8];
#pragma unroll
  for (int t = 0; t < 8; ++t) acc[t] = fzero8();
  gemm_tile(A0, 3 * LDIM, Wb, 3 * LDIM, acc);
  __syncthreads();
  epilogue_lds(acc, b1, true, A1);
  tdm_stage(W2, Wb, LDIM * LDIM);
  stage_wait();

#pragma unroll
  for (int t = 0; t < 8; ++t) acc[t] = fzero8();
  gemm_tile(A1, LDIM, Wb, LDIM, acc);
  __syncthreads();
  epilogue_lds(acc, b2, true, A0);   // reuse A0 (ld = 128)
  tdm_stage(W3, Wb, LDIM * LDIM);
  stage_wait();

#pragma unroll
  for (int t = 0; t < 8; ++t) acc[t] = fzero8();
  gemm_tile(A0, LDIM, Wb, LDIM, acc);

  // scatter-mean numerator: f32 global atomics into agg[dst]
  const int lane = threadIdx.x & 31;
  const int wave = threadIdx.x >> 5;
  const int col0 = lane & 15;
  const int mb = wave * 16 + ((lane >> 4) << 3);
#pragma unroll
  for (int t = 0; t < 8; ++t) {
    const int c = t * 16 + col0;
    const float bv = b3[c];
#pragma unroll
    for (int v = 0; v < 8; ++v) {
      const int m = mb + v;
      if (ebase + m < nE) {
        const int d = dsts[m];
        unsafeAtomicAdd(&agg[(size_t)d * LDIM + c], acc[t][v] + bv);
      }
    }
  }
}

// ---------------------------------------------------------------------------
// Node step: upd_in = [h | agg*inv_deg] (128x256) -> 3-layer MLP -> residual.
// LDS: A0(128x256) | Wb(256x128) | A1(128x128) = 160 KB
// ---------------------------------------------------------------------------
__global__ void node_update_kernel(const unsigned short* h_bf_in,
                                   const float* __restrict__ aggp,
                                   const float* __restrict__ inv_deg,
                                   const unsigned short* W1, const float* b1,
                                   const unsigned short* W2, const float* b2,
                                   const unsigned short* W3, const float* b3,
                                   float* h, unsigned short* h_bf_out, int nrows) {
  extern __shared__ unsigned short smem[];
  unsigned short* A0 = smem;                 // 128*256
  unsigned short* Wb = smem + 128 * 256;     // 256*128
  unsigned short* A1 = Wb + 256 * LDIM;      // 128*128
  const int base = blockIdx.x * LDIM;

  const unsigned a0base = lds_off32(A0);
  const uint4 z4 = {0u, 0u, 0u, 0u};
  for (int i = threadIdx.x; i < LDIM * 16; i += THREADS) {  // h half: async 16B
    const int r = i >> 4, s = i & 15;
    const int row = base + r;
    if (row < nrows)
      async_g2l_b128(a0base + (unsigned)(r * 32 + s) * 16u,
                     h_bf_in + (size_t)row * LDIM + s * 8);
    else
      ((uint4*)A0)[r * 32 + s] = z4;
  }
  for (int i = threadIdx.x; i < LDIM * LDIM; i += THREADS) { // agg half: scaled
    const int r = i >> 7, c = i & 127;
    const int row = base + r;
    float v = 0.0f;
    if (row < nrows) v = aggp[(size_t)row * LDIM + c] * inv_deg[row];
    A0[r * 256 + 128 + c] = f2bf(v);
  }
  tdm_stage(W1, Wb, 2 * LDIM * LDIM);
  stage_wait();

  floatx8 acc[8];
#pragma unroll
  for (int t = 0; t < 8; ++t) acc[t] = fzero8();
  gemm_tile(A0, 2 * LDIM, Wb, 2 * LDIM, acc);
  __syncthreads();
  epilogue_lds(acc, b1, true, A1);
  tdm_stage(W2, Wb, LDIM * LDIM);
  stage_wait();

#pragma unroll
  for (int t = 0; t < 8; ++t) acc[t] = fzero8();
  gemm_tile(A1, LDIM, Wb, LDIM, acc);
  __syncthreads();
  epilogue_lds(acc, b2, true, A0);   // reuse A0 (ld = 128)
  tdm_stage(W3, Wb, LDIM * LDIM);
  stage_wait();

#pragma unroll
  for (int t = 0; t < 8; ++t) acc[t] = fzero8();
  gemm_tile(A0, LDIM, Wb, LDIM, acc);

  const int lane = threadIdx.x & 31;
  const int wave = threadIdx.x >> 5;
  const int col0 = lane & 15;
  const int mb = wave * 16 + ((lane >> 4) << 3);
#pragma unroll
  for (int t = 0; t < 8; ++t) {
    const int c = t * 16 + col0;
    const float bv = b3[c];
#pragma unroll
    for (int v = 0; v < 8; ++v) {
      const int row = base + mb + v;
      if (row < nrows) {
        const size_t idx = (size_t)row * LDIM + c;
        const float hn = h[idx] + acc[t][v] + bv;   // residual
        h[idx] = hn;
        h_bf_out[idx] = f2bf(hn);
      }
    }
  }
}

// ---------------------------------------------------------------------------
// Decoder: 128 -> 128 -> 128 -> 3 (W3 pre-padded bf16 to 128 x 16)
// ---------------------------------------------------------------------------
__global__ void decoder_kernel(const unsigned short* __restrict__ h_bf,
                               const unsigned short* W1, const float* b1,
                               const unsigned short* W2, const float* b2,
                               const unsigned short* W3p, const float* b3,
                               float* __restrict__ out, int nrows) {
  extern __shared__ unsigned short smem[];
  unsigned short* A0 = smem;
  unsigned short* A1 = smem + LDIM * LDIM;
  unsigned short* Wb = smem + 2 * LDIM * LDIM;
  const int base = blockIdx.x * LDIM;

  const unsigned a0base = lds_off32(A0);
  const uint4 z4 = {0u, 0u, 0u, 0u};
  for (int i = threadIdx.x; i < LDIM * 16; i += THREADS) {
    const int r = i >> 4, s = i & 15;
    const int row = base + r;
    if (row < nrows)
      async_g2l_b128(a0base + (unsigned)i * 16u, h_bf + (size_t)row * LDIM + s * 8);
    else
      ((uint4*)A0)[i] = z4;
  }
  tdm_stage(W1, Wb, LDIM * LDIM);
  stage_wait();

  floatx8 acc[8];
#pragma unroll
  for (int t = 0; t < 8; ++t) acc[t] = fzero8();
  gemm_tile(A0, LDIM, Wb, LDIM, acc);
  __syncthreads();
  epilogue_lds(acc, b1, true, A1);
  tdm_stage(W2, Wb, LDIM * LDIM);
  stage_wait();

#pragma unroll
  for (int t = 0; t < 8; ++t) acc[t] = fzero8();
  gemm_tile(A1, LDIM, Wb, LDIM, acc);
  __syncthreads();
  epilogue_lds(acc, b2, true, A0);
  tdm_stage(W3p, Wb, LDIM * 16);
  stage_wait();

  floatx8 a3 = fzero8();
  for (int k = 0; k < LDIM; k += 32) {
    bf16x16 a = load_a_frag(A0, LDIM, k);
    bf16x16 b = load_b_frag(Wb, 16, k, 0);
    a3 = __builtin_amdgcn_wmma_f32_16x16x32_bf16(
        false, a, false, b, (short)0, a3, false, false);
  }
  const int lane = threadIdx.x & 31;
  const int wave = threadIdx.x >> 5;
  const int n = lane & 15;
  const int mb = wave * 16 + ((lane >> 4) << 3);
  if (n < 3) {
#pragma unroll
    for (int v = 0; v < 8; ++v) {
      const int row = base + mb + v;
      if (row < nrows) out[row * 3 + n] = a3[v] + b3[n];
    }
  }
}

// ---------------------------------------------------------------------------
// Small utility kernels
// ---------------------------------------------------------------------------
__global__ void zero_f32_kernel(float* p, int n) {
  const int i = blockIdx.x * blockDim.x + threadIdx.x;
  if (i < n) p[i] = 0.0f;
}
__global__ void count_deg_kernel(const int* __restrict__ edge_index, float* deg, int nE) {
  const int i = blockIdx.x * blockDim.x + threadIdx.x;
  if (i < nE) unsafeAtomicAdd(&deg[edge_index[nE + i]], 1.0f);
}
__global__ void finalize_deg_kernel(float* deg, int n) {
  const int i = blockIdx.x * blockDim.x + threadIdx.x;
  if (i < n) deg[i] = 1.0f / fmaxf(deg[i], 1.0f);
}
__global__ void cvt_bf16_kernel(const float* __restrict__ s, unsigned short* d, int n) {
  const int i = blockIdx.x * blockDim.x + threadIdx.x;
  if (i < n) d[i] = f2bf(s[i]);
}
// d[Kpad x Npad] = zero-padded bf16 of s[K x N]
__global__ void cvt_bf16_pad_kernel(const float* __restrict__ s, unsigned short* d,
                                    int K, int N, int Kpad, int Npad) {
  const int i = blockIdx.x * blockDim.x + threadIdx.x;
  if (i < Kpad * Npad) {
    const int k = i / Npad, c = i % Npad;
    d[i] = (k < K && c < N) ? f2bf(s[k * N + c]) : (unsigned short)0;
  }
}

// ---------------------------------------------------------------------------
// Launcher
// ---------------------------------------------------------------------------
extern "C" void kernel_launch(void* const* d_in, const int* in_sizes, int n_in,
                              void* d_out, int out_size, void* d_ws, size_t ws_size,
                              hipStream_t stream) {
  const float* x         = (const float*)d_in[0];
  const float* edge_attr = (const float*)d_in[1];
  const int*   edge_index= (const int*)d_in[2];
  const float* ne_W1 = (const float*)d_in[3];  const float* ne_b1 = (const float*)d_in[4];
  const float* ne_W2 = (const float*)d_in[5];  const float* ne_b2 = (const float*)d_in[6];
  const float* ne_W3 = (const float*)d_in[7];  const float* ne_b3 = (const float*)d_in[8];
  const float* ee_W1 = (const float*)d_in[9];  const float* ee_b1 = (const float*)d_in[10];
  const float* ee_W2 = (const float*)d_in[11]; const float* ee_b2 = (const float*)d_in[12];
  const float* ee_W3 = (const float*)d_in[13]; const float* ee_b3 = (const float*)d_in[14];
  const float* de_W1 = (const float*)d_in[15]; const float* de_b1 = (const float*)d_in[16];
  const float* de_W2 = (const float*)d_in[17]; const float* de_b2 = (const float*)d_in[18];
  const float* de_W3 = (const float*)d_in[19]; const float* de_b3 = (const float*)d_in[20];
  const float* pe_W1 = (const float*)d_in[21]; const float* pe_b1 = (const float*)d_in[22];
  const float* pe_W2 = (const float*)d_in[23]; const float* pe_b2 = (const float*)d_in[24];
  const float* pe_W3 = (const float*)d_in[25]; const float* pe_b3 = (const float*)d_in[26];
  const float* pn_W1 = (const float*)d_in[27]; const float* pn_b1 = (const float*)d_in[28];
  const float* pn_W2 = (const float*)d_in[29]; const float* pn_b2 = (const float*)d_in[30];
  const float* pn_W3 = (const float*)d_in[31]; const float* pn_b3 = (const float*)d_in[32];

  const int N_ = in_sizes[0] / 7;      // 50000
  const int E_ = in_sizes[1] / 3;      // 800000
  const int S_ = 5;
  const int LL = LDIM * LDIM;          // 16384

  // workspace carve-out (256B aligned)
  char* ws = (char*)d_ws;
  size_t off = 0;
  auto carve = [&](size_t bytes) -> void* {
    void* p = ws + off;
    off += (bytes + 255) & ~(size_t)255;
    return p;
  };
  float*          h       = (float*)carve((size_t)N_ * LDIM * 4);
  unsigned short* h_bf    = (unsigned short*)carve((size_t)N_ * LDIM * 2);
  unsigned short* e_bf    = (unsigned short*)carve((size_t)E_ * LDIM * 2);
  float*          agg     = (float*)carve((size_t)N_ * LDIM * 4);
  float*          inv_deg = (float*)carve((size_t)N_ * 4);
  // bf16 weight mirrors (pre-converted, pre-padded where needed)
  unsigned short* ne1bf = (unsigned short*)carve((size_t)32 * LDIM * 2);
  unsigned short* ne2bf = (unsigned short*)carve((size_t)LL * 2);
  unsigned short* ne3bf = (unsigned short*)carve((size_t)LL * 2);
  unsigned short* ee1bf = (unsigned short*)carve((size_t)32 * LDIM * 2);
  unsigned short* ee2bf = (unsigned short*)carve((size_t)LL * 2);
  unsigned short* ee3bf = (unsigned short*)carve((size_t)LL * 2);
  unsigned short* de1bf = (unsigned short*)carve((size_t)LL * 2);
  unsigned short* de2bf = (unsigned short*)carve((size_t)LL * 2);
  unsigned short* de3bf = (unsigned short*)carve((size_t)LDIM * 16 * 2);
  unsigned short* pe1bf = (unsigned short*)carve((size_t)S_ * 3 * LL * 2);
  unsigned short* pe2bf = (unsigned short*)carve((size_t)S_ * LL * 2);
  unsigned short* pe3bf = (unsigned short*)carve((size_t)S_ * LL * 2);
  unsigned short* pn1bf = (unsigned short*)carve((size_t)S_ * 2 * LL * 2);
  unsigned short* pn2bf = (unsigned short*)carve((size_t)S_ * LL * 2);
  unsigned short* pn3bf = (unsigned short*)carve((size_t)S_ * LL * 2);
  (void)ws_size; (void)n_in; (void)out_size;

  auto cvt = [&](const float* s, unsigned short* d, int n) {
    cvt_bf16_kernel<<<(n + 255) / 256, 256, 0, stream>>>(s, d, n);
  };
  // plain conversions
  cvt(ne_W2, ne2bf, LL);  cvt(ne_W3, ne3bf, LL);
  cvt(ee_W2, ee2bf, LL);  cvt(ee_W3, ee3bf, LL);
  cvt(de_W1, de1bf, LL);  cvt(de_W2, de2bf, LL);
  cvt(pe_W1, pe1bf, S_ * 3 * LL);
  cvt(pe_W2, pe2bf, S_ * LL);
  cvt(pe_W3, pe3bf, S_ * LL);
  cvt(pn_W1, pn1bf, S_ * 2 * LL);
  cvt(pn_W2, pn2bf, S_ * LL);
  cvt(pn_W3, pn3bf, S_ * LL);
  // padded conversions
  cvt_bf16_pad_kernel<<<(32 * LDIM + 255) / 256, 256, 0, stream>>>(ne_W1, ne1bf, 7, LDIM, 32, LDIM);
  cvt_bf16_pad_kernel<<<(32 * LDIM + 255) / 256, 256, 0, stream>>>(ee_W1, ee1bf, 3, LDIM, 32, LDIM);
  cvt_bf16_pad_kernel<<<(LDIM * 16 + 255) / 256, 256, 0, stream>>>(de_W3, de3bf, LDIM, 3, LDIM, 16);

  const int nbN = (N_ + LDIM - 1) / LDIM;
  const int nbE = (E_ + LDIM - 1) / LDIM;
  const size_t SMEM_ENC  = 3 * LL * sizeof(unsigned short);                              //  96 KB
  const size_t SMEM_EDGE = (128 * 384 + 384 * 128 + LL) * sizeof(unsigned short)
                           + 128 * sizeof(int);                                          // ~224 KB
  const size_t SMEM_NODE = (128 * 256 + 256 * 128 + LL) * sizeof(unsigned short);        // 160 KB

  // encoders
  encoder_kernel<7><<<nbN, THREADS, SMEM_ENC, stream>>>(
      x, ne1bf, ne_b1, ne2bf, ne_b2, ne3bf, ne_b3, h, h_bf, N_);
  encoder_kernel<3><<<nbE, THREADS, SMEM_ENC, stream>>>(
      edge_attr, ee1bf, ee_b1, ee2bf, ee_b2, ee3bf, ee_b3, nullptr, e_bf, E_);

  // degree -> 1/max(deg,1)
  zero_f32_kernel<<<(N_ + 255) / 256, 256, 0, stream>>>(inv_deg, N_);
  count_deg_kernel<<<(E_ + 255) / 256, 256, 0, stream>>>(edge_index, inv_deg, E_);
  finalize_deg_kernel<<<(N_ + 255) / 256, 256, 0, stream>>>(inv_deg, N_);

  // message-passing steps
  for (int s = 0; s < S_; ++s) {
    zero_f32_kernel<<<(N_ * LDIM + 255) / 256, 256, 0, stream>>>(agg, N_ * LDIM);
    edge_mlp_kernel<<<nbE, THREADS, SMEM_EDGE, stream>>>(
        edge_index, h_bf, e_bf,
        pe1bf + (size_t)s * 3 * LL, pe_b1 + (size_t)s * LDIM,
        pe2bf + (size_t)s * LL,     pe_b2 + (size_t)s * LDIM,
        pe3bf + (size_t)s * LL,     pe_b3 + (size_t)s * LDIM,
        agg, E_);
    node_update_kernel<<<nbN, THREADS, SMEM_NODE, stream>>>(
        h_bf, agg, inv_deg,
        pn1bf + (size_t)s * 2 * LL, pn_b1 + (size_t)s * LDIM,
        pn2bf + (size_t)s * LL,     pn_b2 + (size_t)s * LDIM,
        pn3bf + (size_t)s * LL,     pn_b3 + (size_t)s * LDIM,
        h, h_bf, N_);
  }

  // decoder
  decoder_kernel<<<nbN, THREADS, SMEM_ENC, stream>>>(
      h_bf, de1bf, de_b1, de2bf, de_b2, de3bf, de_b3, (float*)d_out, N_);
}
